// GumbelTopKGate_39118562132554
// MI455X (gfx1250) — matrix operand order repaired and checked by reading, not compiled, not run on hardware
//
#include <hip/hip_runtime.h>
#include <stdint.h>

// ---------------------------------------------------------------------------
// Gumbel top-K gate for MI455X (gfx1250, wave32, WMMA bf16 path)
//   q,k: [4,8,2048,64] f32   u: [4,2048,2048] f32   out: [4,1,2048,2048] f32
//   logits = (Q' K'^T) / 64 with K-dim = H*D = 512 (mean over heads folded in)
//   mask   = scores >= 16th-largest(scores) per row  (softmax is monotone)
// ---------------------------------------------------------------------------

typedef __attribute__((ext_vector_type(16))) __bf16 v16bf;
typedef __attribute__((ext_vector_type(8)))  float  v8f;

#define B_   4
#define H_   8
#define N_   2048
#define D_   64
#define KD_  512          // H_*D_ : effective GEMM K dimension
#define KTOP 16
#define SSTR 2052         // LDS score row stride (floats); decorrelates banks

union Frag16 { v16bf v; uint4 q[2]; };   // 32 bytes = 16 bf16

__device__ __forceinline__ unsigned short f32_to_bf16(float f) {
    unsigned int x = __float_as_uint(f);
    x += 0x7FFFu + ((x >> 16) & 1u);      // round-to-nearest-even
    return (unsigned short)(x >> 16);
}

// ---- pack k [B,H,N,D] f32  ->  Kp [B,N,H*D] bf16 (row-major per key) ------
__global__ void pack_k_bf16(const float* __restrict__ k,
                            unsigned short* __restrict__ Kp) {
    size_t idx = (size_t)blockIdx.x * blockDim.x + threadIdx.x;
    if (idx >= (size_t)B_ * N_ * KD_) return;
    int    c  = (int)(idx & (KD_ - 1));
    size_t bn = idx >> 9;                 // KD_ = 512
    int    n  = (int)(bn & (N_ - 1));
    int    b  = (int)(bn >> 11);          // N_ = 2048
    int h = c >> 6, d = c & 63;
    float v = k[(((size_t)b * H_ + h) * N_ + n) * D_ + d];
    Kp[idx] = f32_to_bf16(v);
}

// ---- fused GEMM + gumbel + top-16 threshold + mask ------------------------
__global__ __launch_bounds__(256)
void gumbel_topk_gate(const float* __restrict__ q,
                      const float* __restrict__ u,
                      const unsigned short* __restrict__ Kp,
                      float* __restrict__ out) {
    extern __shared__ __align__(16) char smem[];
    unsigned short* Qtile = (unsigned short*)smem;                        // 16 x 512 bf16
    float* scores = (float*)(smem + 16 * KD_ * sizeof(unsigned short));   // 16 x SSTR f32
    float* thresh = scores + 16 * SSTR;                                   // 16 f32

    const int tid  = threadIdx.x;
    const int lane = tid & 31;            // gfx1250 is wave32
    const int wave = tid >> 5;            // 8 waves / block

    const int b       = blockIdx.x >> 7;          // 128 row-tiles per batch
    const int rowBase = (blockIdx.x & 127) << 4;  // 16 rows per block

    // -- stage Q tile into LDS as bf16, gathering heads into K-dim ----------
    for (int idx = tid; idx < 16 * KD_; idx += 256) {
        int row = idx >> 9;
        int c   = idx & (KD_ - 1);
        int h = c >> 6, d = c & 63;
        float v = q[(((size_t)b * H_ + h) * N_ + rowBase + row) * D_ + d];
        Qtile[row * KD_ + c] = f32_to_bf16(v);
    }
    __syncthreads();

    // -- hoist all 16 A fragments (16x32 bf16, ISA 7.12.2 layout) -----------
    //    lanes 0-15:  V0-3 = K 0..7,  V4-7 = K 16..23   (row m = lane)
    //    lanes 16-31: V0-3 = K 8..15, V4-7 = K 24..31   (row m = lane-16)
    const int am    = lane & 15;
    const int abase = (lane < 16) ? 0 : 8;
    Frag16 afrag[16];
    {
        const unsigned short* qrow = Qtile + am * KD_;
        #pragma unroll
        for (int kk = 0; kk < 16; ++kk) {
            afrag[kk].q[0] = *(const uint4*)(qrow + kk * 32 + abase);
            afrag[kk].q[1] = *(const uint4*)(qrow + kk * 32 + abase + 16);
        }
    }

    // -- GEMM: 128 column tiles, 16 per wave; K=512 in 16 WMMA steps --------
    const int ncol = lane & 15;
    const int mofs = (lane >> 4) << 3;    // C/D layout: M = v + 8*(lane/16)
    const int khalf = (lane >> 4) << 3;   // B layout: lanes 0-15 K 0..7,16..23; 16-31 K 8..15,24..31

    for (int i = 0; i < 16; ++i) {
        const int tileBase = (wave + (i << 3)) << 4;
        const unsigned short* kb =
            Kp + ((size_t)b * N_ + tileBase + ncol) * KD_ + khalf;

        if (i < 15)   // warm L2/WGP$ for this wave's next tile
            __builtin_prefetch(Kp + ((size_t)b * N_ + tileBase + 128 + ncol) * KD_, 0, 1);

        v8f acc = {0.f, 0.f, 0.f, 0.f, 0.f, 0.f, 0.f, 0.f};
        #pragma unroll
        for (int kk = 0; kk < 16; ++kk) {
            Frag16 bfrag;
            bfrag.q[0] = *(const uint4*)(kb + kk * 32);       // 8 contiguous bf16
            bfrag.q[1] = *(const uint4*)(kb + kk * 32 + 16);
            acc = __builtin_amdgcn_wmma_f32_16x16x32_bf16(
                false, afrag[kk].v, false, bfrag.v,
                (short)0, acc, false, false);
        }

        // epilogue: scale + gumbel, stash scores in LDS
        #pragma unroll
        for (int v = 0; v < 8; ++v) {
            int   r    = v + mofs;
            int   gcol = tileBase + ncol;
            float uv   = u[((size_t)b * N_ + rowBase + r) * N_ + gcol];
            float g    = -__logf(-__logf(uv + 1e-9f) + 1e-9f);
            scores[r * SSTR + gcol] = acc[v] * (1.0f / 64.0f) + g;
        }
    }
    __syncthreads();

    // -- 16th-largest per row: 16 descending-max passes (2 rows per wave) ---
    for (int rr = 0; rr < 2; ++rr) {
        const int    row  = wave * 2 + rr;
        const float* srow = scores + row * SSTR;
        float bound = __builtin_inff();
        for (int pass = 0; pass < KTOP; ++pass) {
            float m = -__builtin_inff();
            #pragma unroll 4
            for (int j = 0; j < N_ / 32; ++j) {
                float v = srow[lane + (j << 5)];   // bank-conflict-free
                if (v < bound && v > m) m = v;
            }
            #pragma unroll
            for (int off = 16; off >= 1; off >>= 1)
                m = fmaxf(m, __shfl_xor(m, off, 32));
            bound = m;
        }
        if (lane == 0) thresh[row] = bound;
    }
    __syncthreads();

    // -- binary mask, coalesced writes --------------------------------------
    float* orow = out + ((size_t)b * N_ + rowBase) * N_;
    for (int idx = tid; idx < 16 * N_; idx += 256) {
        int row = idx >> 11;
        int col = idx & (N_ - 1);
        orow[(size_t)row * N_ + col] =
            (scores[row * SSTR + col] >= thresh[row]) ? 1.0f : 0.0f;
    }
}

// ---------------------------------------------------------------------------
extern "C" void kernel_launch(void* const* d_in, const int* in_sizes, int n_in,
                              void* d_out, int out_size, void* d_ws, size_t ws_size,
                              hipStream_t stream) {
    const float* q = (const float*)d_in[0];
    const float* k = (const float*)d_in[1];
    const float* u = (const float*)d_in[2];
    unsigned short* Kp = (unsigned short*)d_ws;     // 4*2048*512 bf16 = 8 MB

    const size_t totalK = (size_t)B_ * N_ * KD_;
    pack_k_bf16<<<dim3((unsigned)((totalK + 255) / 256)), dim3(256), 0, stream>>>(k, Kp);

    const size_t smem = (size_t)16 * KD_ * sizeof(unsigned short)   // Q tile
                      + (size_t)16 * SSTR * sizeof(float)           // scores
                      + 16 * sizeof(float);                         // thresholds
    gumbel_topk_gate<<<dim3(B_ * (N_ / 16)), dim3(256), smem, stream>>>(
        q, u, Kp, (float*)d_out);
}
